// StableLSTMCaptioningModel_22943715295436
// MI455X (gfx1250) — matrix operand order repaired and checked by reading, not compiled, not run on hardware
//
#include <hip/hip_runtime.h>

// ---------------------------------------------------------------------------
// Model dims (compile-time, from reference)
// ---------------------------------------------------------------------------
#define BB   32
#define PP   196
#define TT   32
#define EE   512
#define HH   512
#define AA   256
#define VV   30000
#define FEAT 2048
#define MP   (BB * PP)   // 6272

typedef unsigned short bfraw;
typedef __attribute__((ext_vector_type(16))) __bf16 v16bf;
typedef __attribute__((ext_vector_type(8)))  float  v8f;

__device__ __forceinline__ bfraw f2bf(float f) {
    __bf16 h = (__bf16)f;
    return __builtin_bit_cast(bfraw, h);
}
__device__ __forceinline__ __bf16 bfld(bfraw u) {
    return __builtin_bit_cast(__bf16, u);
}
__device__ __forceinline__ float sigmoidf(float x) {
    return 1.0f / (1.0f + __expf(-x));
}

// ---------------------------------------------------------------------------
// Generic helpers
// ---------------------------------------------------------------------------
__global__ void cvt_f32_bf16(const float* __restrict__ in, bfraw* __restrict__ out, int n) {
    int i = blockIdx.x * blockDim.x + threadIdx.x;
    if (i < n) out[i] = f2bf(in[i]);
}

__global__ void zero_u32(unsigned int* __restrict__ p, int nwords) {
    int i = blockIdx.x * blockDim.x + threadIdx.x;
    if (i < nwords) p[i] = 0u;
}

// ---------------------------------------------------------------------------
// WMMA bf16 GEMM:  C[M,N] = A[M,K] @ W[N,K]^T  (+bias) (+=) (clip)
//  A, W bf16 row-major; C f32 with row stride ldc.
//  Block = 256 threads = 8 waves; wave w computes tile (m0, n0=blk*128+w*16).
//  Requires M%16==0, N%16==0, K%32==0 (true for all call sites).
// ---------------------------------------------------------------------------
__global__ void gemm_bf16_wmma(const bfraw* __restrict__ A, const bfraw* __restrict__ W,
                               float* __restrict__ C, const float* __restrict__ bias,
                               int M, int N, int K, int ldc, int accumulate, float clipv) {
    const int lane = threadIdx.x & 31;
    const int wave = threadIdx.x >> 5;
    const int n0 = (blockIdx.x * 8 + wave) * 16;
    const int m0 = blockIdx.y * 16;
    if (n0 >= N || m0 >= M) return;   // wave-uniform exit (EXEC all-1s inside)

    const int arow  = m0 + (lane & 15);
    const int bcol  = n0 + (lane & 15);
    const int ahalf = (lane >> 4) * 8;    // unused; A loads also use bhalf
    const int bhalf = (lane >> 4) * 16;   // B: K offset per lane-half

    const bfraw* Ap = A + (size_t)arow * K;
    const bfraw* Wp = W + (size_t)bcol * K;

    v8f acc = {};
    for (int kb = 0; kb < K; kb += 32) {
        __builtin_prefetch(Ap + kb + 32, 0, 1);   // global_prefetch_b8
        __builtin_prefetch(Wp + kb + 32, 0, 1);
        v16bf a, b;
#pragma unroll
        for (int j = 0; j < 8; ++j) {
            // A 16x32 bf16 fragment: lane = M row, VGPR j -> K = 2j (+16 for lanes 16..31)
            int ka = 2 * j + bhalf + kb;
            a[2 * j]     = bfld(Ap[ka]);
            a[2 * j + 1] = bfld(Ap[ka + 1]);
            // B 32x16 bf16 fragment: lane = N col, VGPR j -> K = 2j (+16 for lanes 16..31)
            int kw = 2 * j + bhalf + kb;
            b[2 * j]     = bfld(Wp[kw]);
            b[2 * j + 1] = bfld(Wp[kw + 1]);
        }
        acc = __builtin_amdgcn_wmma_f32_16x16x32_bf16(false, a, false, b,
                                                      (short)0, acc, false, false);
    }

    const int col = n0 + (lane & 15);
    const float bv = bias ? bias[col] : 0.0f;
#pragma unroll
    for (int r = 0; r < 8; ++r) {
        int row = m0 + r + ((lane >> 4) << 3);   // C/D layout: lanes 16..31 hold M=8..15
        size_t idx = (size_t)row * ldc + col;
        float v = acc[r] + bv;
        if (accumulate) v += C[idx];
        if (clipv > 0.0f) v = fminf(fmaxf(v, -clipv), clipv);
        C[idx] = v;
    }
}

// ---------------------------------------------------------------------------
// Adapter epilogue: per-row ReLU + LayerNorm over E=512; writes f32 + bf16
// One block (256 thr) per row; 2 elements per thread.
// ---------------------------------------------------------------------------
__global__ void adapter_relu_ln(float* __restrict__ Y, bfraw* __restrict__ Yb,
                                const float* __restrict__ g, const float* __restrict__ b) {
    __shared__ float red[256];
    const int r = blockIdx.x;
    const int t = threadIdx.x;
    float* row = Y + (size_t)r * EE;
    float v0 = fmaxf(row[t], 0.0f);
    float v1 = fmaxf(row[t + 256], 0.0f);

    red[t] = v0 + v1;
    __syncthreads();
    for (int s = 128; s > 0; s >>= 1) { if (t < s) red[t] += red[t + s]; __syncthreads(); }
    float mean = red[0] * (1.0f / EE);
    __syncthreads();

    float d0 = v0 - mean, d1 = v1 - mean;
    red[t] = d0 * d0 + d1 * d1;
    __syncthreads();
    for (int s = 128; s > 0; s >>= 1) { if (t < s) red[t] += red[t + s]; __syncthreads(); }
    float inv = rsqrtf(red[0] * (1.0f / EE) + 1e-5f);

    float o0 = d0 * inv * g[t] + b[t];
    float o1 = d1 * inv * g[t + 256] + b[t + 256];
    row[t] = o0;          row[t + 256] = o1;
    Yb[(size_t)r * EE + t] = f2bf(o0);
    Yb[(size_t)r * EE + t + 256] = f2bf(o1);
}

// ---------------------------------------------------------------------------
// Head: LayerNorm(256) + ReLU -> bf16
// ---------------------------------------------------------------------------
__global__ void head_ln_relu(const float* __restrict__ Y, bfraw* __restrict__ Yb,
                             const float* __restrict__ g, const float* __restrict__ b) {
    __shared__ float red[256];
    const int r = blockIdx.x;
    const int t = threadIdx.x;
    float v = Y[r * 256 + t];
    red[t] = v;
    __syncthreads();
    for (int s = 128; s > 0; s >>= 1) { if (t < s) red[t] += red[t + s]; __syncthreads(); }
    float mean = red[0] * (1.0f / 256.0f);
    __syncthreads();
    float d = v - mean;
    red[t] = d * d;
    __syncthreads();
    for (int s = 128; s > 0; s >>= 1) { if (t < s) red[t] += red[t + s]; __syncthreads(); }
    float inv = rsqrtf(red[0] * (1.0f / 256.0f) + 1e-5f);
    float o = d * inv * g[t] + b[t];
    Yb[r * 256 + t] = f2bf(fmaxf(o, 0.0f));
}

// ---------------------------------------------------------------------------
// Additive attention score: e[b,p] = Wf . relu(att1[b,p,:] + att2[b,:]) + bf
// One wave per (b,p) row; wave32 shuffle reduction.
// ---------------------------------------------------------------------------
__global__ void attn_score(const float* __restrict__ att1, const float* __restrict__ att2,
                           const float* __restrict__ Wf, const float* __restrict__ bfv,
                           float* __restrict__ e) {
    const int lane = threadIdx.x & 31;
    const int row  = blockIdx.x * 8 + (threadIdx.x >> 5);
    if (row >= MP) return;
    const int b = row / PP;
    const float* a1 = att1 + (size_t)row * AA;
    const float* a2 = att2 + (size_t)b * AA;
    float s = 0.0f;
    for (int a = lane; a < AA; a += 32)
        s += Wf[a] * fmaxf(a1[a] + a2[a], 0.0f);
    for (int m = 16; m > 0; m >>= 1) s += __shfl_xor(s, m, 32);
    if (lane == 0) e[row] = s + bfv[0];
}

// Softmax over P=196, one block per batch row, in place.
__global__ void softmax_p(float* __restrict__ e) {
    __shared__ float red[256];
    const int b = blockIdx.x, t = threadIdx.x;
    float* row = e + b * PP;
    float v = (t < PP) ? row[t] : -1e30f;
    red[t] = v;
    __syncthreads();
    for (int s = 128; s > 0; s >>= 1) { if (t < s) red[t] = fmaxf(red[t], red[t + s]); __syncthreads(); }
    float mx = red[0];
    __syncthreads();
    float ex = (t < PP) ? __expf(v - mx) : 0.0f;
    red[t] = ex;
    __syncthreads();
    for (int s = 128; s > 0; s >>= 1) { if (t < s) red[t] += red[t + s]; __syncthreads(); }
    float inv = 1.0f / red[0];
    if (t < PP) row[t] = ex * inv;
}

// ctx[b,e] = sum_p alpha[b,p] * enc[b,p,e];  x = [emb_t | ctx] as bf16
__global__ void ctx_assemble(const float* __restrict__ alpha, const float* __restrict__ enc,
                             const bfraw* __restrict__ embs, bfraw* __restrict__ x, int t) {
    int i = blockIdx.x * blockDim.x + threadIdx.x;
    if (i >= BB * EE) return;
    const int b = i >> 9, e = i & 511;
    const float* al = alpha + b * PP;
    const float* ep = enc + ((size_t)b * PP) * EE + e;
    float s = 0.0f;
    for (int p = 0; p < PP; ++p) s += al[p] * ep[(size_t)p * EE];
    x[b * (2 * EE) + EE + e] = f2bf(s);
    x[b * (2 * EE) + e]      = embs[((size_t)b * TT + t) * EE + e];
}

// LSTM cell elementwise: gates [B,4H] (i,f,g,o); updates c (f32), writes h (bf16)
__global__ void lstm_cell(const float* __restrict__ gates, float* __restrict__ c,
                          bfraw* __restrict__ hb) {
    int i = blockIdx.x * blockDim.x + threadIdx.x;
    if (i >= BB * HH) return;
    const int b = i >> 9, j = i & 511;
    const float* g = gates + (size_t)b * (4 * HH);
    float ig = sigmoidf(g[j]);
    float fg = sigmoidf(g[HH + j]);
    float gg = tanhf(g[2 * HH + j]);
    float og = sigmoidf(g[3 * HH + j]);
    float cn = fg * c[i] + ig * gg;
    c[i]  = cn;
    hb[i] = f2bf(og * tanhf(cn));
}

// Embedding gather -> bf16: embs[b,t,e] = emb[captions[b,t], e]
__global__ void embed_gather(const float* __restrict__ emb, const int* __restrict__ cap,
                             bfraw* __restrict__ out) {
    int i = blockIdx.x * blockDim.x + threadIdx.x;
    if (i >= BB * TT * EE) return;
    const int e = i & 511, bt = i >> 9;
    out[i] = f2bf(emb[(size_t)cap[bt] * EE + e]);
}

// ---------------------------------------------------------------------------
// Host launcher
// ---------------------------------------------------------------------------
extern "C" void kernel_launch(void* const* d_in, const int* in_sizes, int n_in,
                              void* d_out, int out_size, void* d_ws, size_t ws_size,
                              hipStream_t stream) {
    (void)in_sizes; (void)n_in; (void)out_size; (void)ws_size;

    const float* features = (const float*)d_in[0];
    const int*   captions = (const int*)d_in[1];
    const float* Wad  = (const float*)d_in[2];
    const float* bad  = (const float*)d_in[3];
    const float* g_ad = (const float*)d_in[4];
    const float* b_ad = (const float*)d_in[5];
    const float* We   = (const float*)d_in[6];
    const float* be   = (const float*)d_in[7];
    const float* Wd   = (const float*)d_in[8];
    const float* bd   = (const float*)d_in[9];
    const float* Wf   = (const float*)d_in[10];
    const float* bfv  = (const float*)d_in[11];
    const float* emb  = (const float*)d_in[12];
    const float* W_ih0 = (const float*)d_in[13];
    const float* W_hh0 = (const float*)d_in[14];
    const float* b_ih0 = (const float*)d_in[15];
    const float* b_hh0 = (const float*)d_in[16];
    const float* W_ih1 = (const float*)d_in[17];
    const float* W_hh1 = (const float*)d_in[18];
    const float* b_ih1 = (const float*)d_in[19];
    const float* b_hh1 = (const float*)d_in[20];
    const float* W1 = (const float*)d_in[21];
    const float* b1 = (const float*)d_in[22];
    const float* g1 = (const float*)d_in[23];
    const float* bb1 = (const float*)d_in[24];
    const float* W2 = (const float*)d_in[25];
    const float* b2 = (const float*)d_in[26];
    float* out = (float*)d_out;

    // Workspace carve-out
    char* ws = (char*)d_ws;
    size_t off = 0;
    auto carve = [&](size_t bytes) -> void* {
        void* p = ws + off;
        off = (off + bytes + 255) & ~(size_t)255;
        return p;
    };
    bfraw* feat_bf = (bfraw*)carve((size_t)MP * FEAT * 2);
    bfraw* Wad_bf  = (bfraw*)carve((size_t)EE * FEAT * 2);
    bfraw* We_bf   = (bfraw*)carve((size_t)AA * EE * 2);
    bfraw* Wd_bf   = (bfraw*)carve((size_t)AA * HH * 2);
    bfraw* Wih0_bf = (bfraw*)carve((size_t)4 * HH * 2 * EE * 2);
    bfraw* Whh0_bf = (bfraw*)carve((size_t)4 * HH * HH * 2);
    bfraw* Wih1_bf = (bfraw*)carve((size_t)4 * HH * HH * 2);
    bfraw* Whh1_bf = (bfraw*)carve((size_t)4 * HH * HH * 2);
    bfraw* W1_bf   = (bfraw*)carve((size_t)(HH / 2) * HH * 2);
    bfraw* W2_bf   = (bfraw*)carve((size_t)VV * (HH / 2) * 2);
    float* enc     = (float*)carve((size_t)MP * EE * 4);
    bfraw* enc_bf  = (bfraw*)carve((size_t)MP * EE * 2);
    float* att1    = (float*)carve((size_t)MP * AA * 4);
    bfraw* embs_bf = (bfraw*)carve((size_t)BB * TT * EE * 2);
    float* att2    = (float*)carve((size_t)BB * AA * 4);
    float* escore  = (float*)carve((size_t)BB * PP * 4);
    bfraw* x_bf    = (bfraw*)carve((size_t)BB * 2 * EE * 2);
    float* gates0  = (float*)carve((size_t)BB * 4 * HH * 4);
    float* gates1  = (float*)carve((size_t)BB * 4 * HH * 4);
    float* y1      = (float*)carve((size_t)BB * (HH / 2) * 4);
    bfraw* y1_bf   = (bfraw*)carve((size_t)BB * (HH / 2) * 2);
    // contiguous LSTM state: c0,c1 f32 then h0,h1 bf16
    float* c0 = (float*)carve((size_t)2 * BB * HH * 4 + (size_t)2 * BB * HH * 2);
    float* c1 = c0 + BB * HH;
    bfraw* h0_bf = (bfraw*)(c1 + BB * HH);
    bfraw* h1_bf = h0_bf + BB * HH;

    const dim3 blk(256);
    auto cvt = [&](const float* src, bfraw* dst, int n) {
        cvt_f32_bf16<<<dim3((n + 255) / 256), blk, 0, stream>>>(src, dst, n);
    };
    auto gemm = [&](const bfraw* A, const bfraw* W, float* C, const float* bias,
                    int M, int N, int K, int ldc, int acc, float clip) {
        dim3 grid((N + 127) / 128, (M + 15) / 16);
        gemm_bf16_wmma<<<grid, blk, 0, stream>>>(A, W, C, bias, M, N, K, ldc, acc, clip);
    };

    // ---- one-time (per launch) conversions ----
    cvt(features, feat_bf, MP * FEAT);
    cvt(Wad,  Wad_bf,  EE * FEAT);
    cvt(We,   We_bf,   AA * EE);
    cvt(Wd,   Wd_bf,   AA * HH);
    cvt(W_ih0, Wih0_bf, 4 * HH * 2 * EE);
    cvt(W_hh0, Whh0_bf, 4 * HH * HH);
    cvt(W_ih1, Wih1_bf, 4 * HH * HH);
    cvt(W_hh1, Whh1_bf, 4 * HH * HH);
    cvt(W1, W1_bf, (HH / 2) * HH);
    cvt(W2, W2_bf, VV * (HH / 2));

    // ---- encoder ----
    gemm(feat_bf, Wad_bf, enc, bad, MP, EE, FEAT, EE, 0, 0.0f);
    adapter_relu_ln<<<dim3(MP), blk, 0, stream>>>(enc, enc_bf, g_ad, b_ad);
    gemm(enc_bf, We_bf, att1, be, MP, AA, EE, AA, 0, 0.0f);
    embed_gather<<<dim3((BB * TT * EE + 255) / 256), blk, 0, stream>>>(emb, captions, embs_bf);

    // zero LSTM state (c0,c1 f32 + h0,h1 bf16, contiguous)
    {
        int nwords = (2 * BB * HH * 4 + 2 * BB * HH * 2) / 4;
        zero_u32<<<dim3((nwords + 255) / 256), blk, 0, stream>>>((unsigned int*)c0, nwords);
    }

    // ---- decode loop (sequential dependence carried by stream order) ----
    for (int t = 0; t < TT; ++t) {
        gemm(h1_bf, Wd_bf, att2, bd, BB, AA, HH, AA, 0, 0.0f);
        attn_score<<<dim3(MP / 8), blk, 0, stream>>>(att1, att2, Wf, bfv, escore);
        softmax_p<<<dim3(BB), blk, 0, stream>>>(escore);
        ctx_assemble<<<dim3((BB * EE + 255) / 256), blk, 0, stream>>>(escore, enc, embs_bf, x_bf, t);

        gemm(x_bf,  Wih0_bf, gates0, b_ih0, BB, 4 * HH, 2 * EE, 4 * HH, 0, 0.0f);
        gemm(h0_bf, Whh0_bf, gates0, b_hh0, BB, 4 * HH, HH,     4 * HH, 1, 0.0f);
        lstm_cell<<<dim3((BB * HH + 255) / 256), blk, 0, stream>>>(gates0, c0, h0_bf);

        gemm(h0_bf, Wih1_bf, gates1, b_ih1, BB, 4 * HH, HH, 4 * HH, 0, 0.0f);
        gemm(h1_bf, Whh1_bf, gates1, b_hh1, BB, 4 * HH, HH, 4 * HH, 1, 0.0f);
        lstm_cell<<<dim3((BB * HH + 255) / 256), blk, 0, stream>>>(gates1, c1, h1_bf);

        gemm(h1_bf, W1_bf, y1, b1, BB, HH / 2, HH, HH / 2, 0, 0.0f);
        head_ln_relu<<<dim3(BB), blk, 0, stream>>>(y1, y1_bf, g1, bb1);
        // logits straight into d_out[b, t, v] with clip(+-10): ldc = T*V, base offset t*V
        gemm(y1_bf, W2_bf, out + (size_t)t * VV, b2, BB, VV, HH / 2, TT * VV, 0, 10.0f);
    }
}
